// BitNetTransformerBlock_11888469475628
// MI455X (gfx1250) — 8-bit matrix operand order repaired and checked by reading, compiled, not run on hardware
//
#include <hip/hip_runtime.h>
#include <hip/hip_bf16.h>
#include <stdint.h>

typedef int   v8i __attribute__((ext_vector_type(8)));
typedef float v8f __attribute__((ext_vector_type(8)));
typedef float v2f __attribute__((ext_vector_type(2)));

#define DIM   2048
#define HEADS 16
#define HDIM  128
#define SEQ   2048
#define BATCH 2
#define TOK   (BATCH*SEQ)

__global__ void layernorm_kernel(const float* __restrict__ x, const float* __restrict__ g,
                                 const float* __restrict__ b, float* __restrict__ out) {
  __shared__ float red[256];
  __shared__ float s_mu, s_rstd;
  const int row = blockIdx.x;
  const int tid = threadIdx.x;
  const float* p = x + (size_t)row * DIM;
  float s = 0.f;
  for (int i = tid; i < DIM; i += 256) s += p[i];
  red[tid] = s; __syncthreads();
  for (int w = 128; w > 0; w >>= 1) { if (tid < w) red[tid] += red[tid + w]; __syncthreads(); }
  if (tid == 0) s_mu = red[0] * (1.f / DIM);
  __syncthreads();
  const float mu = s_mu;
  float vs = 0.f;
  for (int i = tid; i < DIM; i += 256) { float d = p[i] - mu; vs += d * d; }
  red[tid] = vs; __syncthreads();
  for (int w = 128; w > 0; w >>= 1) { if (tid < w) red[tid] += red[tid + w]; __syncthreads(); }
  if (tid == 0) s_rstd = rsqrtf(red[0] * (1.f / DIM) + 1e-5f);
  __syncthreads();
  const float r = s_rstd;
  float* o = out + (size_t)row * DIM;
  for (int i = tid; i < DIM; i += 256) o[i] = (p[i] - mu) * r * g[i] + b[i];
}

__global__ void quant_act_kernel(const float* __restrict__ in, int8_t* __restrict__ q,
                                 float* __restrict__ ascale, int K) {
  __shared__ float red[256];
  const int row = blockIdx.x;
  const int tid = threadIdx.x;
  const float* p = in + (size_t)row * K;
  float amax = 0.f;
  for (int i = tid; i < K; i += 256) amax = fmaxf(amax, fabsf(p[i]));
  red[tid] = amax; __syncthreads();
  for (int w = 128; w > 0; w >>= 1) { if (tid < w) red[tid] = fmaxf(red[tid], red[tid + w]); __syncthreads(); }
  amax = fmaxf(red[0], 1e-5f);
  const float xs = 127.f / amax;
  if (tid == 0) ascale[row] = amax * (1.f / 127.f);
  int8_t* qr = q + (size_t)row * K;
  for (int i = tid; i < K; i += 256) {
    float v = rintf(p[i] * xs);
    v = fminf(fmaxf(v, -128.f), 127.f);
    qr[i] = (int8_t)v;
  }
}

__global__ void absmean_partial_kernel(const float* __restrict__ w, float* __restrict__ partial,
                                       long long n) {
  __shared__ float red[256];
  float s = 0.f;
  for (long long i = blockIdx.x * 256LL + threadIdx.x; i < n; i += (long long)gridDim.x * 256LL)
    s += fabsf(w[i]);
  red[threadIdx.x] = s; __syncthreads();
  for (int w2 = 128; w2 > 0; w2 >>= 1) { if (threadIdx.x < w2) red[threadIdx.x] += red[threadIdx.x + w2]; __syncthreads(); }
  if (threadIdx.x == 0) partial[blockIdx.x] = red[0];
}

__global__ void absmean_final_kernel(const float* __restrict__ partial, int np, float invn,
                                     float* __restrict__ out) {
  __shared__ float red[256];
  float s = 0.f;
  for (int i = threadIdx.x; i < np; i += 256) s += partial[i];
  red[threadIdx.x] = s; __syncthreads();
  for (int w = 128; w > 0; w >>= 1) { if (threadIdx.x < w) red[threadIdx.x] += red[threadIdx.x + w]; __syncthreads(); }
  if (threadIdx.x == 0) out[0] = fmaxf(red[0] * invn, 1e-5f);
}

__global__ void quant_weight_kernel(const float* __restrict__ w, const float* __restrict__ wsp,
                                    int8_t* __restrict__ q, long long n) {
  const float inv = 1.f / (*wsp);
  for (long long i = blockIdx.x * 256LL + threadIdx.x; i < n; i += (long long)gridDim.x * 256LL) {
    float v = rintf(w[i] * inv);
    v = fminf(fmaxf(v, -1.f), 1.f);
    q[i] = (int8_t)v;
  }
}

template <int EPI>
__global__ void gemm_i8_kernel(const int8_t* __restrict__ A, const float* __restrict__ aScale,
                               const int8_t* __restrict__ W, const float* __restrict__ wsPtr,
                               const float* __restrict__ resid, float* __restrict__ out,
                               int M, int N, int K) {
  const int lane = threadIdx.x & 31;
  const int wv   = threadIdx.x >> 5;
  const int hl   = lane >> 4;
  const int l15  = lane & 15;
  const int m0 = blockIdx.x * 64  + (wv & 1) * 32;
  const int n0 = blockIdx.y * 128 + (wv >> 1) * 32;

  v8i acc[2][2] = {};
  const float ws = *wsPtr;

  const int8_t* arow0 = A + (size_t)(m0 + l15) * K + hl * 16;
  const int8_t* arow1 = arow0 + (size_t)16 * K;
  const int8_t* wrow0 = W + (size_t)(n0 + l15) * K + hl * 16;
  const int8_t* wrow1 = wrow0 + (size_t)16 * K;

  for (int kb = 0; kb < K; kb += 64) {
    v8i af[2], bf[2];
    {
      const int8_t* ap = arow0 + kb;
      int2 d0 = *(const int2*)(ap);
      int2 d1 = *(const int2*)(ap + 8);
      int2 d2 = *(const int2*)(ap + 32);
      int2 d3 = *(const int2*)(ap + 40);
      af[0] = (v8i){d0.x, d0.y, d1.x, d1.y, d2.x, d2.y, d3.x, d3.y};
      ap = arow1 + kb;
      d0 = *(const int2*)(ap);
      d1 = *(const int2*)(ap + 8);
      d2 = *(const int2*)(ap + 32);
      d3 = *(const int2*)(ap + 40);
      af[1] = (v8i){d0.x, d0.y, d1.x, d1.y, d2.x, d2.y, d3.x, d3.y};
      const int8_t* wp = wrow0 + kb;
      int4 c0 = *(const int4*)(wp);
      int4 c1 = *(const int4*)(wp + 32);
      bf[0] = (v8i){c0.x, c0.y, c0.z, c0.w, c1.x, c1.y, c1.z, c1.w};
      wp = wrow1 + kb;
      c0 = *(const int4*)(wp);
      c1 = *(const int4*)(wp + 32);
      bf[1] = (v8i){c0.x, c0.y, c0.z, c0.w, c1.x, c1.y, c1.z, c1.w};
      if (kb + 64 < K) {
        __builtin_prefetch(wrow0 + kb + 64, 0, 0);
        __builtin_prefetch(wrow1 + kb + 64, 0, 0);
      }
    }
    acc[0][0] = __builtin_amdgcn_wmma_i32_16x16x64_iu8(true, af[0], true, bf[0], acc[0][0], false, false);
    acc[0][1] = __builtin_amdgcn_wmma_i32_16x16x64_iu8(true, af[0], true, bf[1], acc[0][1], false, false);
    acc[1][0] = __builtin_amdgcn_wmma_i32_16x16x64_iu8(true, af[1], true, bf[0], acc[1][0], false, false);
    acc[1][1] = __builtin_amdgcn_wmma_i32_16x16x64_iu8(true, af[1], true, bf[1], acc[1][1], false, false);
  }

#pragma unroll
  for (int mt = 0; mt < 2; ++mt)
#pragma unroll
    for (int nt = 0; nt < 2; ++nt)
#pragma unroll
      for (int j = 0; j < 8; ++j) {
        const int row = m0 + mt * 16 + j + hl * 8;
        const int col = n0 + nt * 16 + l15;
        float v = (float)acc[mt][nt][j] * ws * aScale[row];
        if (EPI == 1) v += resid[(size_t)row * N + col];
        if (EPI == 2) v = 0.5f * v * (1.f + erff(v * 0.70710678118654752f));
        out[(size_t)row * N + col] = v;
      }
}

__global__ void __launch_bounds__(128, 1)
attention_kernel(const float* __restrict__ qkv, float* __restrict__ obuf) {
  __shared__ float sP[4 * 256];
  const int lane = threadIdx.x & 31;
  const int wv   = threadIdx.x >> 5;
  const int hl   = lane >> 4;
  const int l15  = lane & 15;
  const int bh = blockIdx.y;
  const int b  = bh >> 4;
  const int h  = bh & 15;
  const int qt = blockIdx.x * 4 + wv;
  const int qtok0 = b * SEQ + qt * 16;
  const float scale = 0.08838834764831845f;

  v2f qf[32];
  const float* qrow = qkv + (size_t)(qtok0 + l15) * (3 * DIM) + h * HDIM + 2 * hl;
#pragma unroll
  for (int f = 0; f < 32; ++f) { qf[f].x = qrow[4 * f] * scale; qf[f].y = qrow[4 * f + 1] * scale; }

  v8f oacc[8];
#pragma unroll
  for (int d = 0; d < 8; ++d) oacc[d] = (v8f){};
  float m_i[8], l_i[8];
#pragma unroll
  for (int j = 0; j < 8; ++j) { m_i[j] = -3.4e38f; l_i[j] = 0.f; }

  for (int kt = 0; kt < SEQ / 16; ++kt) {
    const int ktok0 = b * SEQ + kt * 16;
    const float* krow = qkv + (size_t)(ktok0 + l15) * (3 * DIM) + DIM + h * HDIM + 2 * hl;

    v8f s0 = {}, s1 = {};
#pragma unroll
    for (int half2 = 0; half2 < 2; ++half2) {
      v2f kf[16];
#pragma unroll
      for (int f = 0; f < 16; ++f) {
        kf[f].x = krow[4 * (half2 * 16 + f)];
        kf[f].y = krow[4 * (half2 * 16 + f) + 1];
      }
#pragma unroll
      for (int f = 0; f < 8; ++f) {
        s0 = __builtin_amdgcn_wmma_f32_16x16x4_f32(false, qf[half2 * 16 + 2 * f],     false, kf[2 * f],     (short)0, s0, false, false);
        s1 = __builtin_amdgcn_wmma_f32_16x16x4_f32(false, qf[half2 * 16 + 2 * f + 1], false, kf[2 * f + 1], (short)0, s1, false, false);
      }
    }
    v8f sacc = s0 + s1;

    float corr[8];
#pragma unroll
    for (int j = 0; j < 8; ++j) {
      float t = sacc[j];
      t = fmaxf(t, __shfl_xor(t, 8, 16));
      t = fmaxf(t, __shfl_xor(t, 4, 16));
      t = fmaxf(t, __shfl_xor(t, 2, 16));
      t = fmaxf(t, __shfl_xor(t, 1, 16));
      const float mn = fmaxf(m_i[j], t);
      corr[j] = expf(m_i[j] - mn);
      m_i[j] = mn;
      float pv = expf(sacc[j] - mn);
      float rs = pv;
      rs += __shfl_xor(rs, 8, 16);
      rs += __shfl_xor(rs, 4, 16);
      rs += __shfl_xor(rs, 2, 16);
      rs += __shfl_xor(rs, 1, 16);
      l_i[j] = l_i[j] * corr[j] + rs;
      sP[wv * 256 + (j + 8 * hl) * 16 + l15] = pv;
    }
    asm volatile("s_wait_dscnt 0" ::: "memory");
#pragma unroll
    for (int d = 0; d < 8; ++d)
#pragma unroll
      for (int j = 0; j < 8; ++j) oacc[d][j] *= corr[j];

    v2f pf[4];
#pragma unroll
    for (int g = 0; g < 4; ++g) {
      pf[g].x = sP[wv * 256 + l15 * 16 + 4 * g + 2 * hl];
      pf[g].y = sP[wv * 256 + l15 * 16 + 4 * g + 2 * hl + 1];
    }

    const float* vbase = qkv + (size_t)ktok0 * (3 * DIM) + 2 * DIM + h * HDIM;
#pragma unroll
    for (int db = 0; db < 2; ++db) {
      v2f vf[4][4];
#pragma unroll
      for (int dd = 0; dd < 4; ++dd) {
        const float* vcol = vbase + (db * 4 + dd) * 16 + l15;
#pragma unroll
        for (int g = 0; g < 4; ++g) {
          vf[dd][g].x = vcol[(size_t)(4 * g + 2 * hl)     * (3 * DIM)];
          vf[dd][g].y = vcol[(size_t)(4 * g + 2 * hl + 1) * (3 * DIM)];
        }
      }
#pragma unroll
      for (int g = 0; g < 4; ++g)
#pragma unroll
        for (int dd = 0; dd < 4; ++dd)
          oacc[db * 4 + dd] = __builtin_amdgcn_wmma_f32_16x16x4_f32(
              false, pf[g], false, vf[dd][g], (short)0, oacc[db * 4 + dd], false, false);
    }
  }
#pragma unroll
  for (int d = 0; d < 8; ++d)
#pragma unroll
    for (int j = 0; j < 8; ++j) {
      const int tok = qtok0 + j + 8 * hl;
      const int col = h * HDIM + d * 16 + l15;
      obuf[(size_t)tok * DIM + col] = oacc[d][j] / l_i[j];
    }
}

extern "C" void kernel_launch(void* const* d_in, const int* in_sizes, int n_in,
                              void* d_out, int out_size, void* d_ws, size_t ws_size,
                              hipStream_t stream) {
  (void)in_sizes; (void)n_in; (void)out_size; (void)ws_size;
  const float* x      = (const float*)d_in[0];
  const float* w_qkv  = (const float*)d_in[1];
  const float* w_proj = (const float*)d_in[2];
  const float* w_fc1  = (const float*)d_in[3];
  const float* w_fc2  = (const float*)d_in[4];
  const float* g1 = (const float*)d_in[5];
  const float* b1 = (const float*)d_in[6];
  const float* g2 = (const float*)d_in[7];
  const float* b2 = (const float*)d_in[8];
  float* out = (float*)d_out;

  char* ws = (char*)d_ws;
  int8_t* wq_qkv  = (int8_t*)(ws + 0ULL);
  int8_t* wq_proj = (int8_t*)(ws + 12582912ULL);
  int8_t* wq_fc1  = (int8_t*)(ws + 16777216ULL);
  int8_t* wq_fc2  = (int8_t*)(ws + 33554432ULL);
  float*  scal    = (float*)(ws + 50331648ULL);
  float*  partial = (float*)(ws + 50331904ULL);
  float*  lnbuf   = (float*)(ws + 50397184ULL);
  int8_t* aq      = (int8_t*)(ws + 83951616ULL);
  float*  ascale  = (float*)(ws + 117506048ULL);
  float*  bigbuf  = (float*)(ws + 117571584ULL);
  float*  obuf    = (float*)(ws + 251789312ULL);

  struct WQ { const float* w; int8_t* q; long long n; int si; };
  const WQ wl[4] = {
      {w_qkv,  wq_qkv,  6144LL * 2048, 0},
      {w_proj, wq_proj, 2048LL * 2048, 1},
      {w_fc1,  wq_fc1,  8192LL * 2048, 2},
      {w_fc2,  wq_fc2,  2048LL * 8192, 3}};
  for (int i = 0; i < 4; ++i) {
    absmean_partial_kernel<<<1024, 256, 0, stream>>>(wl[i].w, partial, wl[i].n);
    absmean_final_kernel<<<1, 256, 0, stream>>>(partial, 1024, 1.f / (float)wl[i].n, scal + wl[i].si);
    quant_weight_kernel<<<4096, 256, 0, stream>>>(wl[i].w, scal + wl[i].si, wl[i].q, wl[i].n);
  }

  layernorm_kernel<<<TOK, 256, 0, stream>>>(x, g1, b1, lnbuf);
  quant_act_kernel<<<TOK, 256, 0, stream>>>(lnbuf, aq, ascale, DIM);
  gemm_i8_kernel<0><<<dim3(TOK / 64, (3 * DIM) / 128), 256, 0, stream>>>(
      aq, ascale, wq_qkv, scal + 0, nullptr, bigbuf, TOK, 3 * DIM, DIM);
  attention_kernel<<<dim3(SEQ / 64, BATCH * HEADS), 128, 0, stream>>>(bigbuf, obuf);
  quant_act_kernel<<<TOK, 256, 0, stream>>>(obuf, aq, ascale, DIM);
  gemm_i8_kernel<1><<<dim3(TOK / 64, DIM / 128), 256, 0, stream>>>(
      aq, ascale, wq_proj, scal + 1, x, out, TOK, DIM, DIM);

  layernorm_kernel<<<TOK, 256, 0, stream>>>(out, g2, b2, lnbuf);
  quant_act_kernel<<<TOK, 256, 0, stream>>>(lnbuf, aq, ascale, DIM);
  gemm_i8_kernel<2><<<dim3(TOK / 64, (4 * DIM) / 128), 256, 0, stream>>>(
      aq, ascale, wq_fc1, scal + 2, nullptr, bigbuf, TOK, 4 * DIM, DIM);
  quant_act_kernel<<<TOK, 256, 0, stream>>>(bigbuf, aq, ascale, 4 * DIM);
  gemm_i8_kernel<1><<<dim3(TOK / 64, DIM / 128), 256, 0, stream>>>(
      aq, ascale, wq_fc2, scal + 3, out, out, TOK, DIM, 4 * DIM);
}
